// MultiHeadAttention_86655260164934
// MI455X (gfx1250) — matrix operand order repaired and checked by reading, not compiled, not run on hardware
//
#include <hip/hip_runtime.h>
#include <hip/hip_bf16.h>

// ---------------------------------------------------------------------------
// MHA forward for B=2, S=2048, E=1024, H=16, DK=64 on gfx1250 (MI455X).
// All matmuls via v_wmma_f32_16x16x32_bf16.  All LDS fragment loads are
// 16-byte ds_load_b128 (LDS tiles stored so each fragment is contiguous).
// ---------------------------------------------------------------------------

typedef __attribute__((ext_vector_type(16))) __bf16 v16bf;
typedef __attribute__((ext_vector_type(8)))  __bf16 v8bf;
typedef __attribute__((ext_vector_type(4)))  __bf16 v4bf;
typedef __attribute__((ext_vector_type(8)))  float  v8f;

#define CB 2
#define CS 2048
#define CE 1024
#define CH 16
#define CDK 64

// Native f32->bf16 (RNE) — lowers to v_cvt_*bf16_f32 on gfx1250.
__device__ __forceinline__ __bf16 f2bf(float f) { return (__bf16)f; }

__device__ __forceinline__ v4bf pk4(float4 v) {
    v4bf p = { f2bf(v.x), f2bf(v.y), f2bf(v.z), f2bf(v.w) };
    return p;
}

// Two 16-byte LDS loads -> one 16-element bf16 fragment (register concat).
__device__ __forceinline__ v16bf ld2x8(const __bf16* p0, const __bf16* p1) {
    v8bf lo = *(const v8bf*)p0;
    v8bf hi = *(const v8bf*)p1;
    return __builtin_shufflevector(lo, hi, 0, 1, 2, 3, 4, 5, 6, 7,
                                           8, 9, 10, 11, 12, 13, 14, 15);
}

// ---------------------------------------------------------------------------
// Tiled GEMM: Y = X @ W^T + bias.  X:[M,1024], W:[N,1024] row-major.
// Block = 128 threads (4 waves).  Block tile 128(M) x 64(N), K-step 32.
// Each wave computes 32x64 (8 WMMAs per K-step).
// headSplit=1: scatter output to [B,H,S,DK];  headSplit=0: row-major [M,N].
// ---------------------------------------------------------------------------
__global__ __launch_bounds__(128)
void proj_gemm(const float* __restrict__ X, const float* __restrict__ W,
               const float* __restrict__ bias, float* __restrict__ Y,
               int headSplit)
{
    __shared__ __bf16 As[128][40];   // [m][k], 80B row stride (16B aligned)
    __shared__ __bf16 Bs[64][40];    // [n][k]  (natural W row layout)

    const int tid  = threadIdx.x;
    const int lane = tid & 31;
    const int w    = tid >> 5;
    const int half = lane >> 4;
    const int nl   = lane & 15;

    const int m0 = blockIdx.y * 128;
    const int n0 = blockIdx.x * 64;
    const int Kd = CE;

    v8f acc[2][4] = {};

    for (int k0 = 0; k0 < Kd; k0 += 32) {
        // stage A tile 128x32 (f32 -> bf16): 1024 float4s, 8 per thread
        #pragma unroll
        for (int it = 0; it < 8; ++it) {
            int idx = tid + it * 128;
            int r   = idx >> 3;
            int c4  = (idx & 7) * 4;
            float4 v = *(const float4*)(X + (size_t)(m0 + r) * Kd + k0 + c4);
            *(v4bf*)&As[r][c4] = pk4(v);
        }
        // stage B tile 64x32, N-major: Bs[n][k] = W[n0+n][k0+k]
        #pragma unroll
        for (int it = 0; it < 4; ++it) {
            int idx = tid + it * 128;
            int r   = idx >> 3;
            int c4  = (idx & 7) * 4;
            float4 v = *(const float4*)(W + (size_t)(n0 + r) * Kd + k0 + c4);
            *(v4bf*)&Bs[r][c4] = pk4(v);
        }
        __syncthreads();

        // A fragments: row fixed, k = half*16 + e contiguous
        v16bf a[2];
        #pragma unroll
        for (int mt = 0; mt < 2; ++mt) {
            const __bf16* row = &As[w * 32 + mt * 16 + nl][0];
            a[mt] = ld2x8(row + half * 16, row + half * 16 + 8);
        }
        // B fragments: n fixed, k = half*16 + e contiguous
        #pragma unroll
        for (int nb = 0; nb < 4; ++nb) {
            const __bf16* row = &Bs[nb * 16 + nl][0];
            v16bf bfr = ld2x8(row + half * 16, row + half * 16 + 8);
            #pragma unroll
            for (int mt = 0; mt < 2; ++mt)
                acc[mt][nb] = __builtin_amdgcn_wmma_f32_16x16x32_bf16(
                    false, a[mt], false, bfr, (short)0, acc[mt][nb],
                    false, false);
        }
        __syncthreads();
    }

    // epilogue: bias (4 loads), one row base per (mt,r), offset stores
    float bv[4];
    #pragma unroll
    for (int nb = 0; nb < 4; ++nb) bv[nb] = bias[n0 + nb * 16 + nl];

    if (headSplit) {
        const int hh = n0 >> 6;                 // whole N-tile = one head
        #pragma unroll
        for (int mt = 0; mt < 2; ++mt) {
            #pragma unroll
            for (int r = 0; r < 8; ++r) {
                int gm = m0 + w * 32 + mt * 16 + r + half * 8;
                int bb = gm >> 11;
                int ss = gm & (CS - 1);
                float* rowp = Y + (((size_t)bb * CH + hh) * CS + ss) * CDK + nl;
                #pragma unroll
                for (int nb = 0; nb < 4; ++nb)
                    rowp[nb * 16] = acc[mt][nb][r] + bv[nb];
            }
        }
    } else {
        #pragma unroll
        for (int mt = 0; mt < 2; ++mt) {
            #pragma unroll
            for (int r = 0; r < 8; ++r) {
                int gm = m0 + w * 32 + mt * 16 + r + half * 8;
                float* rowp = Y + (size_t)gm * CE + n0 + nl;
                #pragma unroll
                for (int nb = 0; nb < 4; ++nb)
                    rowp[nb * 16] = acc[mt][nb][r] + bv[nb];
            }
        }
    }
}

// ---------------------------------------------------------------------------
// Flash-attention: one wave per 16-row Q tile; 4 waves/block share (b,h) and
// cooperatively stage 32x64 K (key-major) and V (dk-major, transposed) tiles
// into LDS as bf16.  Online softmax; all fragments via ds_load_b128.
// ---------------------------------------------------------------------------
__global__ __launch_bounds__(128)
void attn_kernel(const float* __restrict__ Q, const float* __restrict__ Kb,
                 const float* __restrict__ Vb, const float* __restrict__ bias,
                 const int* __restrict__ mask, float* __restrict__ ctx)
{
    __shared__ __bf16 Kt[32][72];       // [key][dk], 144B stride
    __shared__ __bf16 Vt[64][40];       // [dk][key], transposed, 80B stride
    __shared__ __bf16 Pl[4][16][40];    // per-wave P tile, [row][key]

    const int tid  = threadIdx.x;
    const int lane = tid & 31;
    const int w    = tid >> 5;
    const int half = lane >> 4;
    const int nl   = lane & 15;

    const int tile = blockIdx.x * 4 + w;      // 4096 q-tiles total
    const int qt   = tile & (CS / 16 - 1);
    const int bh   = tile >> 7;
    const int b    = bh >> 4;
    const int h    = bh & (CH - 1);
    const int q0   = qt * 16;

    const float* Qp = Q  + (size_t)bh * CS * CDK;
    const float* Kp = Kb + (size_t)bh * CS * CDK;
    const float* Vp = Vb + (size_t)bh * CS * CDK;

    // Q fragments (dk 0..31 and 32..63), pre-scaled by 1/sqrt(DK)=0.125.
    // A-layout runs of 8 are contiguous in global memory: load float4 pairs.
    v16bf aq0, aq1;
    {
        const float* qrow = Qp + (size_t)(q0 + nl) * CDK;
        #pragma unroll
        for (int g = 0; g < 2; ++g) {          // runs at k=half*8, 16+half*8
            int kbase = g * 16 + half * 8;
            float4 u0 = *(const float4*)(qrow + kbase);
            float4 u1 = *(const float4*)(qrow + kbase + 4);
            float4 u2 = *(const float4*)(qrow + 32 + kbase);
            float4 u3 = *(const float4*)(qrow + 32 + kbase + 4);
            aq0[g * 8 + 0] = f2bf(0.125f * u0.x); aq0[g * 8 + 1] = f2bf(0.125f * u0.y);
            aq0[g * 8 + 2] = f2bf(0.125f * u0.z); aq0[g * 8 + 3] = f2bf(0.125f * u0.w);
            aq0[g * 8 + 4] = f2bf(0.125f * u1.x); aq0[g * 8 + 5] = f2bf(0.125f * u1.y);
            aq0[g * 8 + 6] = f2bf(0.125f * u1.z); aq0[g * 8 + 7] = f2bf(0.125f * u1.w);
            aq1[g * 8 + 0] = f2bf(0.125f * u2.x); aq1[g * 8 + 1] = f2bf(0.125f * u2.y);
            aq1[g * 8 + 2] = f2bf(0.125f * u2.z); aq1[g * 8 + 3] = f2bf(0.125f * u2.w);
            aq1[g * 8 + 4] = f2bf(0.125f * u3.x); aq1[g * 8 + 5] = f2bf(0.125f * u3.y);
            aq1[g * 8 + 6] = f2bf(0.125f * u3.z); aq1[g * 8 + 7] = f2bf(0.125f * u3.w);
        }
    }

    float rmax[8], rsum[8];
    int   mq[8];
    v8f   o[4] = {};
    #pragma unroll
    for (int r = 0; r < 8; ++r) {
        rmax[r] = -__builtin_inff();
        rsum[r] = 0.0f;
        mq[r]   = mask[b * CS + q0 + r + half * 8];
    }
    const float LOG2E = 1.4426950408889634f;

    for (int k0 = 0; k0 < CS; k0 += 32) {
        // stage K (key-major) and V (dk-major / transposed) tiles
        #pragma unroll
        for (int it = 0; it < 4; ++it) {
            int idx = tid + it * 128;          // 512 float4s
            int r   = idx >> 4;                // key row
            int c4  = (idx & 15) * 4;          // dk col
            float4 kv4 = *(const float4*)(Kp + (size_t)(k0 + r) * CDK + c4);
            *(v4bf*)&Kt[r][c4] = pk4(kv4);
            float4 vv4 = *(const float4*)(Vp + (size_t)(k0 + r) * CDK + c4);
            Vt[c4 + 0][r] = f2bf(vv4.x); Vt[c4 + 1][r] = f2bf(vv4.y);
            Vt[c4 + 2][r] = f2bf(vv4.z); Vt[c4 + 3][r] = f2bf(vv4.w);
        }
        if (k0 + 32 < CS) {   // prefetch next tiles (global_prefetch_b8)
            __builtin_prefetch(Kp + (size_t)(k0 + 32) * CDK + lane * 16, 0, 3);
            __builtin_prefetch(Vp + (size_t)(k0 + 32) * CDK + lane * 16, 0, 3);
        }
        __syncthreads();

        // scores S = (Q/8) @ K^T for two 16-key column blocks
        v8f s[2] = {};
        #pragma unroll
        for (int cb = 0; cb < 2; ++cb) {
            const __bf16* krow = &Kt[cb * 16 + nl][0];
            v16bf bk0 = ld2x8(krow + half * 8, krow + 16 + half * 8);
            s[cb] = __builtin_amdgcn_wmma_f32_16x16x32_bf16(
                false, aq0, false, bk0, (short)0, s[cb], false, false);
            v16bf bk1 = ld2x8(krow + 32 + half * 8, krow + 32 + 16 + half * 8);
            s[cb] = __builtin_amdgcn_wmma_f32_16x16x32_bf16(
                false, aq1, false, bk1, (short)0, s[cb], false, false);
        }

        // bias + mask (C-layout: row = q0+r+half*8, col = k0+cb*16+nl)
        #pragma unroll
        for (int cb = 0; cb < 2; ++cb) {
            int col = k0 + cb * 16 + nl;
            int mk  = mask[b * CS + col];
            const float* bp = bias + ((size_t)b * CS + q0 + half * 8) * CS + col;
            #pragma unroll
            for (int r = 0; r < 8; ++r) {
                float v = s[cb][r] + bp[(size_t)r * CS];
                s[cb][r] = (mk == 0 || mq[r] == 0) ? -1e9f : v;
            }
        }

        // online softmax (row spread over 16 lanes of one half: xor 1,2,4,8)
        #pragma unroll
        for (int r = 0; r < 8; ++r) {
            float t = fmaxf(s[0][r], s[1][r]);
            t = fmaxf(t, __shfl_xor(t, 1));
            t = fmaxf(t, __shfl_xor(t, 2));
            t = fmaxf(t, __shfl_xor(t, 4));
            t = fmaxf(t, __shfl_xor(t, 8));
            float mnew = fmaxf(rmax[r], t);
            float corr = exp2f((rmax[r] - mnew) * LOG2E);
            rmax[r] = mnew;
            float p0 = exp2f((s[0][r] - mnew) * LOG2E);
            float p1 = exp2f((s[1][r] - mnew) * LOG2E);
            s[0][r] = p0; s[1][r] = p1;
            float ts = p0 + p1;
            ts += __shfl_xor(ts, 1);
            ts += __shfl_xor(ts, 2);
            ts += __shfl_xor(ts, 4);
            ts += __shfl_xor(ts, 8);
            rsum[r] = rsum[r] * corr + ts;
            #pragma unroll
            for (int nb = 0; nb < 4; ++nb) o[nb][r] *= corr;
        }

        // P: C-layout -> LDS -> A-layout (16x32 bf16)
        #pragma unroll
        for (int cb = 0; cb < 2; ++cb)
            #pragma unroll
            for (int r = 0; r < 8; ++r)
                Pl[w][r + half * 8][cb * 16 + nl] = f2bf(s[cb][r]);
        asm volatile("s_wait_dscnt 0x0" ::: "memory");
        const __bf16* prow = &Pl[w][nl][0];
        v16bf pa = ld2x8(prow + half * 8, prow + 16 + half * 8);

        // O += P @ V  (4 dk blocks of 16); V is dk-major so frag is contiguous
        #pragma unroll
        for (int nb = 0; nb < 4; ++nb) {
            const __bf16* vrow = &Vt[nb * 16 + nl][0];
            v16bf bv = ld2x8(vrow + half * 8, vrow + 16 + half * 8);
            o[nb] = __builtin_amdgcn_wmma_f32_16x16x32_bf16(
                false, pa, false, bv, (short)0, o[nb], false, false);
        }
        __syncthreads();
    }

    // normalize and store context [B,S,E] with col = h*64 + d
    #pragma unroll
    for (int r = 0; r < 8; ++r) {
        float inv = 1.0f / rsum[r];
        int row = q0 + r + half * 8;
        float* rowp = ctx + ((size_t)b * CS + row) * CE + h * CDK + nl;
        #pragma unroll
        for (int nb = 0; nb < 4; ++nb)
            rowp[nb * 16] = o[nb][r] * inv;
    }
}

// ---------------------------------------------------------------------------
extern "C" void kernel_launch(void* const* d_in, const int* in_sizes, int n_in,
                              void* d_out, int out_size, void* d_ws, size_t ws_size,
                              hipStream_t stream) {
    const float* x         = (const float*)d_in[0];
    const float* kv        = (const float*)d_in[1];
    const int*   mask      = (const int*)  d_in[2];
    const float* attn_bias = (const float*)d_in[3];
    const float* WQ_w = (const float*)d_in[4];
    const float* WQ_b = (const float*)d_in[5];
    const float* WK_w = (const float*)d_in[6];
    const float* WK_b = (const float*)d_in[7];
    const float* WV_w = (const float*)d_in[8];
    const float* WV_b = (const float*)d_in[9];
    const float* WO_w = (const float*)d_in[10];
    const float* WO_b = (const float*)d_in[11];
    float* out = (float*)d_out;

    const size_t NQKV = (size_t)CB * CH * CS * CDK;   // 4,194,304 floats
    float* ws  = (float*)d_ws;
    float* Qb  = ws;
    float* Kb  = ws + NQKV;
    float* Vb  = ws + 2 * NQKV;
    float* ctx = ws + 3 * NQKV;

    dim3 block(128);
    dim3 gridp(CE / 64, (CB * CS) / 128);   // (16, 32)

    proj_gemm<<<gridp, block, 0, stream>>>(x,  WQ_w, WQ_b, Qb, 1);
    proj_gemm<<<gridp, block, 0, stream>>>(kv, WK_w, WK_b, Kb, 1);
    proj_gemm<<<gridp, block, 0, stream>>>(kv, WV_w, WV_b, Vb, 1);

    attn_kernel<<<dim3((CB * CH * (CS / 16)) / 4), block, 0, stream>>>(
        Qb, Kb, Vb, attn_bias, mask, ctx);

    proj_gemm<<<gridp, block, 0, stream>>>(ctx, WO_w, WO_b, out, 0);
}
